// InterModalityUpdate_53687091200655
// MI455X (gfx1250) — matrix operand order repaired and checked by reading, not compiled, not run on hardware
//
#include <hip/hip_runtime.h>

// ---------------------------------------------------------------------------
// InterModalityUpdate for MI455X (gfx1250, wave32, WMMA bf16)
// B=32, N=512, D=1024, H=8, HD=128
// ---------------------------------------------------------------------------

typedef __bf16 bf16;
typedef bf16  v16bf __attribute__((ext_vector_type(16)));
typedef bf16  v8bf  __attribute__((ext_vector_type(8)));
typedef bf16  v4bf  __attribute__((ext_vector_type(4)));
typedef float v8f   __attribute__((ext_vector_type(8)));
typedef float v4f   __attribute__((ext_vector_type(4)));

#define BN   16384   // B*N rows
#define DIM  1024
#define DIM3 3072
#define NSEQ 512
#define NHEAD 8
#define HD   128

__device__ __forceinline__ v8f wmma_bf16(v16bf a, v16bf b, v8f c) {
  return __builtin_amdgcn_wmma_f32_16x16x32_bf16(false, a, false, b,
                                                 (short)0, c, false, false);
}

// Build a 16-element bf16 fragment from two 8-element (16-byte) chunks.
__device__ __forceinline__ v16bf ldfrag(const bf16* p0, const bf16* p1) {
  v8bf lo = *(const v8bf*)p0;
  v8bf hi = *(const v8bf*)p1;
  return __builtin_shufflevector(lo, hi, 0, 1, 2, 3, 4, 5, 6, 7,
                                 8, 9, 10, 11, 12, 13, 14, 15);
}

// ---------------------------------------------------------------------------
// Kernel 1: out(bf16)[BN,3072] = relu( A(f32)[BN,1024] @ W(f32)[1024,3072] + b )
// block 256 threads = 8 waves; tile 128x128, K-step 32.
// ---------------------------------------------------------------------------
__global__ __launch_bounds__(256) void proj_gemm_kernel(
    const float* __restrict__ A, const float* __restrict__ W,
    const float* __restrict__ bias, bf16* __restrict__ out) {
  __shared__ __align__(16) bf16 As[128][32];  // [m][k]
  __shared__ __align__(16) bf16 Bs[128][32];  // [n][k] (transposed)

  const int t    = threadIdx.x;
  const int lane = t & 31;
  const int w    = t >> 5;
  const int half = (lane >> 4) & 1;
  const int l16  = lane & 15;
  const int n0 = blockIdx.x * 128;
  const int m0 = blockIdx.y * 128;
  const int wm = w >> 2;  // 0..1 -> 64 rows
  const int wn = w & 3;   // 0..3 -> 32 cols

  const int arow = t >> 3;        // 0..31
  const int akc  = (t & 7) * 4;   // 0..28
  const int brow = t >> 3;        // k 0..31
  const int bnc  = (t & 7) * 16;  // 0..112

  v8f zero = {};
  v8f acc[4][2];
#pragma unroll
  for (int mt = 0; mt < 4; ++mt)
#pragma unroll
    for (int nt = 0; nt < 2; ++nt) acc[mt][nt] = zero;

  for (int kt = 0; kt < 32; ++kt) {
    const int k0 = kt * 32;
    __syncthreads();
    // ---- stage A tile (fp32 -> bf16)
#pragma unroll
    for (int p = 0; p < 4; ++p) {
      const int r = arow + p * 32;
      v4f f = *(const v4f*)(A + (size_t)(m0 + r) * DIM + k0 + akc);
      v4bf hbf;
      hbf[0] = (bf16)f[0]; hbf[1] = (bf16)f[1];
      hbf[2] = (bf16)f[2]; hbf[3] = (bf16)f[3];
      *(v4bf*)&As[r][akc] = hbf;
    }
    // ---- stage B tile transposed (fp32 -> bf16)
    {
      const float* wp = W + (size_t)(k0 + brow) * DIM3 + n0 + bnc;
#pragma unroll
      for (int c = 0; c < 4; ++c) {
        v4f f = *(const v4f*)(wp + c * 4);
#pragma unroll
        for (int i = 0; i < 4; ++i) Bs[bnc + c * 4 + i][brow] = (bf16)f[i];
      }
    }
    if (kt + 1 < 32) {  // gfx1250 global_prefetch_b8 for the next tiles
      __builtin_prefetch(A + (size_t)(m0 + arow) * DIM + k0 + 32 + akc, 0, 1);
      __builtin_prefetch(W + (size_t)(k0 + 32 + brow) * DIM3 + n0 + bnc, 0, 1);
    }
    __syncthreads();

    // ---- fragments + WMMA
    v16bf af[4];
#pragma unroll
    for (int mt = 0; mt < 4; ++mt) {
      const bf16* rp = &As[wm * 64 + mt * 16 + l16][0];
      const int s0 = half ? 8 : 0;  // lanes<16: K{0..7,16..23}; >=16: {8..15,24..31}
      af[mt] = ldfrag(rp + s0, rp + s0 + 16);
    }
#pragma unroll
    for (int nt = 0; nt < 2; ++nt) {
      const bf16* rp = &Bs[wn * 32 + nt * 16 + l16][0];
      v16bf bfr = ldfrag(rp + half * 8, rp + half * 8 + 16);
#pragma unroll
      for (int mt = 0; mt < 4; ++mt)
        acc[mt][nt] = wmma_bf16(af[mt], bfr, acc[mt][nt]);
    }
  }

  // ---- epilogue: bias + relu, bf16 store
#pragma unroll
  for (int mt = 0; mt < 4; ++mt) {
    const int rbase = m0 + wm * 64 + mt * 16 + (half ? 8 : 0);
#pragma unroll
    for (int nt = 0; nt < 2; ++nt) {
      const int c = n0 + wn * 32 + nt * 16 + l16;
      const float bb = bias[c];
#pragma unroll
      for (int j = 0; j < 8; ++j) {
        float vv = acc[mt][nt][j] + bb;
        vv = fmaxf(vv, 0.0f);
        out[(size_t)(rbase + j) * DIM3 + c] = (bf16)vv;
      }
    }
  }
}

// ---------------------------------------------------------------------------
// Kernel 2: dual cross-attention (flash-style streaming softmax).
// grid = (N/128, B*H, 2); block = 256 (8 waves); wave owns 16 query rows.
// z==0: Q=v_query, K=q_key, V=q_val -> v_update
// z==1: Q=q_query, K=v_key, V=v_val -> q_update
// ---------------------------------------------------------------------------
__global__ __launch_bounds__(256) void attn_kernel(
    const bf16* __restrict__ vt, const bf16* __restrict__ qt,
    bf16* __restrict__ vu, bf16* __restrict__ qu) {
  __shared__ __align__(16) bf16 Ks[32][128];     // [key][hd]
  __shared__ __align__(16) bf16 Vt[128][32];     // [hd][key] (transposed)
  __shared__ __align__(16) bf16 Ps[8][16][32];   // per-wave P re-layout scratch

  const int t    = threadIdx.x;
  const int lane = t & 31;
  const int w    = t >> 5;
  const int half = (lane >> 4) & 1;
  const int l16  = lane & 15;
  const int b = blockIdx.y >> 3;
  const int h = blockIdx.y & 7;
  const int qrow0 = blockIdx.x * 128 + w * 16;

  const bf16 *Qb, *Kb, *Vb;
  bf16* Ob;
  if (blockIdx.z == 0) { Qb = vt + DIM; Kb = qt; Vb = qt + 2 * DIM; Ob = vu; }
  else                 { Qb = qt + DIM; Kb = vt; Vb = vt + 2 * DIM; Ob = qu; }

  const float scale = 0.08838834764831845f;  // 1/sqrt(128)

  // Q fragments (16 rows x 128 hd) resident in registers.
  const bf16* qp = Qb + (size_t)(b * NSEQ + qrow0 + l16) * DIM3 + h * HD;
  v16bf qf[4];
#pragma unroll
  for (int kk = 0; kk < 4; ++kk) {
    const int s0 = kk * 32 + (half ? 8 : 0);
    qf[kk] = ldfrag(qp + s0, qp + s0 + 16);
  }

  v8f zero = {};
  v8f acc[8];
#pragma unroll
  for (int nt = 0; nt < 8; ++nt) acc[nt] = zero;
  float runM[8], runL[8];
#pragma unroll
  for (int j = 0; j < 8; ++j) { runM[j] = -1e30f; runL[j] = 0.0f; }

  const int key = t >> 3;        // 0..31
  const int seg = (t & 7) * 16;  // 0..112

  for (int kc = 0; kc < 16; ++kc) {  // 512 keys, 32 per chunk
    __syncthreads();
    {  // cooperative K / V staging (V transposed)
      const bf16* kp = Kb + (size_t)(b * NSEQ + kc * 32 + key) * DIM3 + h * HD + seg;
      v8bf ka = *(const v8bf*)kp;
      v8bf kb = *(const v8bf*)(kp + 8);
      *(v8bf*)&Ks[key][seg]     = ka;
      *(v8bf*)&Ks[key][seg + 8] = kb;
      const bf16* vp = Vb + (size_t)(b * NSEQ + kc * 32 + key) * DIM3 + h * HD + seg;
      v8bf va = *(const v8bf*)vp;
      v8bf vb = *(const v8bf*)(vp + 8);
#pragma unroll
      for (int i = 0; i < 8; ++i) {
        Vt[seg + i][key]     = va[i];
        Vt[seg + 8 + i][key] = vb[i];
      }
    }
    __syncthreads();

    // ---- S = Q @ K^T (two 16x16 score tiles, contraction 128 = 4 steps)
    v8f s0 = {}, s1 = {};
#pragma unroll
    for (int kk = 0; kk < 4; ++kk) {
      const bf16* r0 = &Ks[l16][kk * 32 + half * 8];
      const bf16* r1 = &Ks[16 + l16][kk * 32 + half * 8];
      v16bf b0 = ldfrag(r0, r0 + 16);
      v16bf b1 = ldfrag(r1, r1 + 16);
      s0 = wmma_bf16(qf[kk], b0, s0);
      s1 = wmma_bf16(qf[kk], b1, s1);
    }

    // ---- online softmax (rows live in C-layout VGPR j + half*8)
#pragma unroll
    for (int j = 0; j < 8; ++j) {
      const float a0 = s0[j] * scale;
      const float a1 = s1[j] * scale;
      float cm = fmaxf(a0, a1);
      cm = fmaxf(cm, __shfl_xor(cm, 1, 32));
      cm = fmaxf(cm, __shfl_xor(cm, 2, 32));
      cm = fmaxf(cm, __shfl_xor(cm, 4, 32));
      cm = fmaxf(cm, __shfl_xor(cm, 8, 32));
      const float nm   = fmaxf(runM[j], cm);
      const float corr = __expf(runM[j] - nm);
      const float p0 = __expf(a0 - nm);
      const float p1 = __expf(a1 - nm);
      float rs = p0 + p1;
      rs += __shfl_xor(rs, 1, 32);
      rs += __shfl_xor(rs, 2, 32);
      rs += __shfl_xor(rs, 4, 32);
      rs += __shfl_xor(rs, 8, 32);
      runL[j] = runL[j] * corr + rs;
      runM[j] = nm;
#pragma unroll
      for (int nt = 0; nt < 8; ++nt) acc[nt][j] *= corr;
      const int row = j + (half ? 8 : 0);
      Ps[w][row][l16]      = (bf16)p0;
      Ps[w][row][16 + l16] = (bf16)p1;
    }

    // ---- O += P(16x32) @ V(32x128), P re-read in A-fragment layout
    {
      const bf16* pp = &Ps[w][l16][half ? 8 : 0];
      v16bf pf = ldfrag(pp, pp + 16);
#pragma unroll
      for (int nt = 0; nt < 8; ++nt) {
        const bf16* vrp = &Vt[nt * 16 + l16][half * 8];
        v16bf bv = ldfrag(vrp, vrp + 16);
        acc[nt] = wmma_bf16(pf, bv, acc[nt]);
      }
    }
  }

  // ---- epilogue: normalize, bf16 store into update buffer [B*N, D]
#pragma unroll
  for (int j = 0; j < 8; ++j) {
    const float inv = 1.0f / runL[j];
    const int r = b * NSEQ + qrow0 + j + (half ? 8 : 0);
#pragma unroll
    for (int nt = 0; nt < 8; ++nt) {
      const int c = h * HD + nt * 16 + l16;
      Ob[(size_t)r * DIM + c] = (bf16)(acc[nt][j] * inv);
    }
  }
}

// ---------------------------------------------------------------------------
// Kernel 3: out(f32)[BN,1024] = relu( [X(f32)[BN,1024] | U(bf16)[BN,1024]]
//                                     @ W(f32)[2048,1024] + b )
// ---------------------------------------------------------------------------
__global__ __launch_bounds__(256) void outproj_kernel(
    const float* __restrict__ X, const bf16* __restrict__ U,
    const float* __restrict__ W, const float* __restrict__ bias,
    float* __restrict__ out) {
  __shared__ __align__(16) bf16 As[128][32];
  __shared__ __align__(16) bf16 Bs[128][32];

  const int t    = threadIdx.x;
  const int lane = t & 31;
  const int w    = t >> 5;
  const int half = (lane >> 4) & 1;
  const int l16  = lane & 15;
  const int n0 = blockIdx.x * 128;
  const int m0 = blockIdx.y * 128;
  const int wm = w >> 2;
  const int wn = w & 3;

  const int arow = t >> 3;
  const int akc  = (t & 7) * 4;
  const int brow = t >> 3;
  const int bnc  = (t & 7) * 16;

  v8f zero = {};
  v8f acc[4][2];
#pragma unroll
  for (int mt = 0; mt < 4; ++mt)
#pragma unroll
    for (int nt = 0; nt < 2; ++nt) acc[mt][nt] = zero;

  for (int kt = 0; kt < 64; ++kt) {  // K = 2048
    const int k0 = kt * 32;
    __syncthreads();
    if (k0 < DIM) {  // first half of concat: fp32 input, convert
#pragma unroll
      for (int p = 0; p < 4; ++p) {
        const int r = arow + p * 32;
        v4f f = *(const v4f*)(X + (size_t)(m0 + r) * DIM + k0 + akc);
        v4bf hbf;
        hbf[0] = (bf16)f[0]; hbf[1] = (bf16)f[1];
        hbf[2] = (bf16)f[2]; hbf[3] = (bf16)f[3];
        *(v4bf*)&As[r][akc] = hbf;
      }
    } else {  // second half: bf16 attention update
#pragma unroll
      for (int p = 0; p < 4; ++p) {
        const int r = arow + p * 32;
        v4bf u = *(const v4bf*)(U + (size_t)(m0 + r) * DIM + (k0 - DIM) + akc);
        *(v4bf*)&As[r][akc] = u;
      }
    }
    {
      const float* wp = W + (size_t)(k0 + brow) * DIM + n0 + bnc;
#pragma unroll
      for (int c = 0; c < 4; ++c) {
        v4f f = *(const v4f*)(wp + c * 4);
#pragma unroll
        for (int i = 0; i < 4; ++i) Bs[bnc + c * 4 + i][brow] = (bf16)f[i];
      }
    }
    if (kt + 1 < 64)
      __builtin_prefetch(W + (size_t)(k0 + 32 + brow) * DIM + n0 + bnc, 0, 1);
    __syncthreads();

    v16bf af[4];
#pragma unroll
    for (int mt = 0; mt < 4; ++mt) {
      const bf16* rp = &As[wm * 64 + mt * 16 + l16][0];
      const int s0 = half ? 8 : 0;
      af[mt] = ldfrag(rp + s0, rp + s0 + 16);
    }
#pragma unroll
    for (int nt = 0; nt < 2; ++nt) {
      const bf16* rp = &Bs[wn * 32 + nt * 16 + l16][0];
      v16bf bfr = ldfrag(rp + half * 8, rp + half * 8 + 16);
#pragma unroll
      for (int mt = 0; mt < 4; ++mt)
        acc[mt][nt] = wmma_bf16(af[mt], bfr, acc[mt][nt]);
    }
  }

#pragma unroll
  for (int mt = 0; mt < 4; ++mt) {
    const int rbase = m0 + wm * 64 + mt * 16 + (half ? 8 : 0);
#pragma unroll
    for (int nt = 0; nt < 2; ++nt) {
      const int c = n0 + wn * 32 + nt * 16 + l16;
      const float bb = bias[c];
#pragma unroll
      for (int j = 0; j < 8; ++j) {
        float vv = acc[mt][nt][j] + bb;
        out[(size_t)(rbase + j) * DIM + c] = fmaxf(vv, 0.0f);
      }
    }
  }
}

// ---------------------------------------------------------------------------
extern "C" void kernel_launch(void* const* d_in, const int* in_sizes, int n_in,
                              void* d_out, int out_size, void* d_ws, size_t ws_size,
                              hipStream_t stream) {
  (void)in_sizes; (void)n_in; (void)out_size; (void)ws_size;
  const float* v   = (const float*)d_in[0];
  const float* q   = (const float*)d_in[1];
  const float* Wv  = (const float*)d_in[2];
  const float* bv  = (const float*)d_in[3];
  const float* Wq  = (const float*)d_in[4];
  const float* bq  = (const float*)d_in[5];
  const float* Wvo = (const float*)d_in[6];
  const float* bvo = (const float*)d_in[7];
  const float* Wqo = (const float*)d_in[8];
  const float* bqo = (const float*)d_in[9];

  // workspace layout (bf16): v_tran[BN,3072] | q_tran[BN,3072] |
  //                          v_update[BN,1024] | q_update[BN,1024]
  bf16* vt = (bf16*)d_ws;
  bf16* qt = vt + (size_t)BN * DIM3;
  bf16* vu = qt + (size_t)BN * DIM3;
  bf16* qu = vu + (size_t)BN * DIM;

  float* out_v = (float*)d_out;
  float* out_q = out_v + (size_t)BN * DIM;

  // 1) input projections + ReLU -> bf16
  proj_gemm_kernel<<<dim3(DIM3 / 128, BN / 128), 256, 0, stream>>>(v, Wv, bv, vt);
  proj_gemm_kernel<<<dim3(DIM3 / 128, BN / 128), 256, 0, stream>>>(q, Wq, bq, qt);

  // 2) dual cross-attention (both directions via grid.z)
  attn_kernel<<<dim3(NSEQ / 128, 32 * NHEAD, 2), 256, 0, stream>>>(vt, qt, vu, qu);

  // 3) output projections (concat handled inside) + ReLU -> fp32 d_out
  outproj_kernel<<<dim3(DIM / 128, BN / 128), 256, 0, stream>>>(v, vu, Wvo, bvo, out_v);
  outproj_kernel<<<dim3(DIM / 128, BN / 128), 256, 0, stream>>>(q, qu, Wqo, bqo, out_q);
}
